// RelativeMultiHeadAttn_88021059764226
// MI455X (gfx1250) — matrix operand order repaired and checked by reading, not compiled, not run on hardware
//
#include <hip/hip_runtime.h>
#include <stdint.h>

// ---------------------------------------------------------------------------
// Types for CDNA5 WMMA (wave32, 16x16x32 bf16 -> f32 accumulate)
// ---------------------------------------------------------------------------
typedef __attribute__((ext_vector_type(16))) __bf16 v16bf;
typedef __attribute__((ext_vector_type(8)))  float  v8f;

union Frag16 {
    v16bf  v;
    __bf16 h[16];
    uint4  q[2];
};
static_assert(sizeof(Frag16) == 32, "Frag16 must be 32 bytes");

__device__ __forceinline__ v8f wmma_bf16(const Frag16& a, const Frag16& b, v8f c) {
    return __builtin_amdgcn_wmma_f32_16x16x32_bf16(false, a.v, false, b.v,
                                                   (short)0, c, false, false);
}

#define QLEN   1024
#define BSZ    4
#define NH     16
#define DH     64
#define DMODEL 1024

// Convert 8 contiguous fp32 -> 8 bf16 (into LDS)
__device__ __forceinline__ void cvt8(const float* __restrict__ src, __bf16* dst) {
    float4 f0 = *(const float4*)src;
    float4 f1 = *(const float4*)(src + 4);
    dst[0] = (__bf16)f0.x; dst[1] = (__bf16)f0.y; dst[2] = (__bf16)f0.z; dst[3] = (__bf16)f0.w;
    dst[4] = (__bf16)f1.x; dst[5] = (__bf16)f1.y; dst[6] = (__bf16)f1.z; dst[7] = (__bf16)f1.w;
}

// ---------------------------------------------------------------------------
// Kernel 1: projection GEMM  C[m,h] = sum_k A[m,k] * W[h,k]
//   mode 0: A = w [4096,1024], W = qkv_w [3072,1024]
//           epilogue: Q(+r_w_bias)->Qw, Q(+r_r_bias)->Qr, K->Kt[b,n,j,d],
//                     V->Vt[b,n,d,j] (transposed for PV WMMA B-frags)
//   mode 1: A = r [1024,1024], W = r_w [1024,1024] -> RK[n,j,d]
// Block tile: 64(M) x 128(N) x 32(K); 256 threads = 8 waves, 32x32 per wave.
// ---------------------------------------------------------------------------
__global__ __launch_bounds__(256) void proj_gemm_kernel(
    const float* __restrict__ A, const float* __restrict__ W, int Kdim, int mode,
    const float* __restrict__ rwb, const float* __restrict__ rrb,
    __bf16* __restrict__ Qw, __bf16* __restrict__ Qr,
    __bf16* __restrict__ Kt, __bf16* __restrict__ Vt, __bf16* __restrict__ RK)
{
    __shared__ __bf16 As[64 * 32];
    __shared__ __bf16 Ws[128 * 32];

    const int tid  = threadIdx.x;
    const int wave = tid >> 5, lane = tid & 31;
    const int hsel = lane >> 4, lrow = lane & 15;
    const int kb   = hsel * 8;    // A-frag K base
    const int koff = hsel * 16;   // unused; B-frags use kb too
    const int mw   = wave & 1, nw = wave >> 1;
    const int m0   = blockIdx.y * 64, n0 = blockIdx.x * 128;

    v8f acc[2][2] = {};

    for (int k0 = 0; k0 < Kdim; k0 += 32) {
        __syncthreads();
        {   // load A tile 64x32 (fp32 -> bf16)
            int r = tid >> 2, c = (tid & 3) * 8;
            cvt8(A + (size_t)(m0 + r) * Kdim + k0 + c, As + r * 32 + c);
        }
#pragma unroll
        for (int p = 0; p < 2; ++p) {   // load W tile 128x32
            int r = p * 64 + (tid >> 2), c = (tid & 3) * 8;
            cvt8(W + (size_t)(n0 + r) * Kdim + k0 + c, Ws + r * 32 + c);
        }
        __syncthreads();

        Frag16 af[2], bf[2];
#pragma unroll
        for (int mi = 0; mi < 2; ++mi) {
            const __bf16* p = As + (mw * 32 + mi * 16 + lrow) * 32;
            af[mi].q[0] = *(const uint4*)(p + kb);
            af[mi].q[1] = *(const uint4*)(p + kb + 16);
        }
#pragma unroll
        for (int ni = 0; ni < 2; ++ni) {
            const __bf16* p = Ws + (nw * 32 + ni * 16 + lrow) * 32;
            bf[ni].q[0] = *(const uint4*)(p + kb);
            bf[ni].q[1] = *(const uint4*)(p + kb + 16);
        }
#pragma unroll
        for (int mi = 0; mi < 2; ++mi)
#pragma unroll
            for (int ni = 0; ni < 2; ++ni)
                acc[mi][ni] = wmma_bf16(af[mi], bf[ni], acc[mi][ni]);
    }

    // epilogue: scatter per element (C layout: M = e + 8*hsel, N = lrow)
#pragma unroll
    for (int mi = 0; mi < 2; ++mi)
#pragma unroll
    for (int ni = 0; ni < 2; ++ni)
#pragma unroll
    for (int e = 0; e < 8; ++e) {
        int gm = m0 + mw * 32 + mi * 16 + e + 8 * hsel;
        int gn = n0 + nw * 32 + ni * 16 + lrow;
        float val = acc[mi][ni][e];
        if (mode == 0) {
            int i = gm >> 2, b = gm & 3;             // w rows are (i, b)
            int part = gn >> 10, hh = gn & 1023;
            int n = hh >> 6, d = hh & 63;
            size_t bn = (size_t)b * NH + n;
            if (part == 0) {
                Qw[(bn * QLEN + i) * DH + d] = (__bf16)(val + rwb[hh]);
                Qr[(bn * QLEN + i) * DH + d] = (__bf16)(val + rrb[hh]);
            } else if (part == 1) {
                Kt[(bn * QLEN + i) * DH + d] = (__bf16)val;
            } else {
                Vt[(bn * DH + d) * QLEN + i] = (__bf16)val;   // transposed V
            }
        } else {
            int n = gn >> 6, d = gn & 63;
            RK[((size_t)n * QLEN + gm) * DH + d] = (__bf16)val;
        }
    }
}

// ---------------------------------------------------------------------------
// Kernel 2: fused relative attention, flash style.
// Grid: (qlen/16, bsz, n_head); 128 threads = 4 waves.
// Phase A: BD_orig strip rows [i0 .. i0+16] x 1024 into LDS (bf16).
// Phase B: each wave owns j-tiles {wave, wave+4, ...}, online softmax,
//          PV via K=16 zero-padded bf16 WMMA; waves merged through LDS.
// rel_shift: BDs[i,j] = BD[i, j+L-1-i] (j<=i), 0 (j==i+1), BD[i+1, j-i-2] (j>i+1)
// ---------------------------------------------------------------------------
__global__ __launch_bounds__(128) void attn_kernel(
    const __bf16* __restrict__ Qw, const __bf16* __restrict__ Qr,
    const __bf16* __restrict__ Kt, const __bf16* __restrict__ Vt,
    const __bf16* __restrict__ RK, __bf16* __restrict__ AV)
{
    __shared__ __bf16 BD[17 * 1024];       // 34 KB
    __shared__ float  Pld[4][256];         //  4 KB (per-wave P scratch)
    __shared__ float  Ocomb[4][16 * 64];   // 16 KB
    __shared__ float  Stats[4][16][2];     // .5 KB

    const int i0 = blockIdx.x * 16;
    const int b  = blockIdx.y, n = blockIdx.z;
    const int tid = threadIdx.x, wave = tid >> 5, lane = tid & 31;
    const int hsel = lane >> 4, lrow = lane & 15;
    const int kb = hsel * 8, koff = hsel * 16;

    const size_t bn = (size_t)b * NH + n;
    const __bf16* Qwp = Qw + bn * QLEN * DH;
    const __bf16* Qrp = Qr + bn * QLEN * DH;
    const __bf16* Kp  = Kt + bn * QLEN * DH;
    const __bf16* Vp  = Vt + bn * DH * QLEN;
    const __bf16* RKp = RK + (size_t)n * QLEN * DH;

    // Preload Q A-fragments (16 rows; aqr2 = rows i0+16.., clamped)
    Frag16 aqw[2], aqr[2], aqr2[2];
    int r2 = i0 + 16 + lrow; if (r2 > QLEN - 1) r2 = QLEN - 1;
#pragma unroll
    for (int ks = 0; ks < 2; ++ks) {
        const __bf16* p1 = Qwp + (i0 + lrow) * DH + ks * 32 + kb;
        aqw[ks].q[0] = *(const uint4*)p1;
        aqw[ks].q[1] = *(const uint4*)(p1 + 16);
        const __bf16* p2 = Qrp + (i0 + lrow) * DH + ks * 32 + kb;
        aqr[ks].q[0] = *(const uint4*)p2;
        aqr[ks].q[1] = *(const uint4*)(p2 + 16);
        const __bf16* p3 = Qrp + r2 * DH + ks * 32 + kb;
        aqr2[ks].q[0] = *(const uint4*)p3;
        aqr2[ks].q[1] = *(const uint4*)(p3 + 16);
    }

    // ---- Phase A: BD strip (17 rows x 1024 cols) ----
    for (int t = wave; t < 64; t += 4) {
        int c0 = t * 16;
        Frag16 br[2];
#pragma unroll
        for (int ks = 0; ks < 2; ++ks) {
            const __bf16* p = RKp + (c0 + lrow) * DH + ks * 32 + kb;
            br[ks].q[0] = *(const uint4*)p;
            br[ks].q[1] = *(const uint4*)(p + 16);
        }
        v8f a0 = {};
        a0 = wmma_bf16(aqr[0], br[0], a0);
        a0 = wmma_bf16(aqr[1], br[1], a0);
#pragma unroll
        for (int e = 0; e < 8; ++e)
            BD[(e + 8 * hsel) * 1024 + c0 + lrow] = (__bf16)a0[e];
        v8f a1 = {};
        a1 = wmma_bf16(aqr2[0], br[0], a1);
        a1 = wmma_bf16(aqr2[1], br[1], a1);
        if (hsel == 0) BD[16 * 1024 + c0 + lrow] = (__bf16)a1[0];  // row i0+16 only
    }
    __syncthreads();

    // ---- Phase B: flash attention over j-tiles ----
    float mrun[8], lrun[8];
    v8f O[4] = {};
#pragma unroll
    for (int e = 0; e < 8; ++e) { mrun[e] = -1e30f; lrun[e] = 0.f; }

    for (int t = wave; t < 64; t += 4) {
        int j0 = t * 16;
        Frag16 bk[2];
#pragma unroll
        for (int ks = 0; ks < 2; ++ks) {
            const __bf16* p = Kp + (j0 + lrow) * DH + ks * 32 + kb;
            bk[ks].q[0] = *(const uint4*)p;
            bk[ks].q[1] = *(const uint4*)(p + 16);
        }
        v8f acc = {};
        acc = wmma_bf16(aqw[0], bk[0], acc);
        acc = wmma_bf16(aqw[1], bk[1], acc);

        float pex[8];
#pragma unroll
        for (int e = 0; e < 8; ++e) {
            int m = e + 8 * hsel;
            int i = i0 + m;
            int j = j0 + lrow;
            float bd;
            if (j <= i)          bd = (float)BD[m * 1024 + (j + (QLEN - 1) - i)];
            else if (j == i + 1) bd = 0.f;
            else                 bd = (float)BD[(m + 1) * 1024 + (j - i - 2)];
            float s = (acc[e] + bd) * 0.125f;            // 1/sqrt(64)
            float mx = s;
#pragma unroll
            for (int msk = 1; msk < 16; msk <<= 1)
                mx = fmaxf(mx, __shfl_xor(mx, msk, 32)); // row max over 16 lanes
            float newm = fmaxf(mrun[e], mx);
            float cf = __expf(mrun[e] - newm);
            float pe = __expf(s - newm);
            float ps = pe;
#pragma unroll
            for (int msk = 1; msk < 16; msk <<= 1)
                ps += __shfl_xor(ps, msk, 32);           // row sum
            lrun[e] = lrun[e] * cf + ps;
            mrun[e] = newm;
#pragma unroll
            for (int dt = 0; dt < 4; ++dt) O[dt][e] *= cf;
            pex[e] = pe;
        }
        // stage P (C-layout) -> re-read as A-frag (zero-padded K=16..31)
#pragma unroll
        for (int e = 0; e < 8; ++e)
            Pld[wave][(e + 8 * hsel) * 16 + lrow] = pex[e];
        Frag16 pf;
#pragma unroll
        for (int c = 0; c < 8; ++c) pf.h[c] = (__bf16)Pld[wave][lrow * 16 + kb + c];
#pragma unroll
        for (int c = 8; c < 16; ++c) pf.h[c] = (__bf16)0.f;

#pragma unroll
        for (int dt = 0; dt < 4; ++dt) {
            Frag16 vf;
            {   // B-frag elems 0..7 cover K=8*hsel..+7 (real j's); elems 8..15 (K=16..31) are the zero pad
                const __bf16* p = Vp + (size_t)(dt * 16 + lrow) * QLEN + j0 + kb;
                vf.q[0] = *(const uint4*)p;
                vf.q[1] = make_uint4(0, 0, 0, 0);
            }
            O[dt] = wmma_bf16(pf, vf, O[dt]);
        }
    }

    // ---- merge the 4 per-wave partial softmax states ----
#pragma unroll
    for (int dt = 0; dt < 4; ++dt)
#pragma unroll
        for (int e = 0; e < 8; ++e)
            Ocomb[wave][(e + 8 * hsel) * 64 + dt * 16 + lrow] = O[dt][e];
    if (lrow == 0) {
#pragma unroll
        for (int e = 0; e < 8; ++e) {
            int m = e + 8 * hsel;
            Stats[wave][m][0] = mrun[e];
            Stats[wave][m][1] = lrun[e];
        }
    }
    __syncthreads();

    for (int kk = 0; kk < 8; ++kk) {
        int pos = tid + 128 * kk;            // 1024 outputs / 128 threads
        int row = pos >> 6, col = pos & 63;
        float M = -1e30f;
#pragma unroll
        for (int wv = 0; wv < 4; ++wv) M = fmaxf(M, Stats[wv][row][0]);
        float L = 0.f, val = 0.f;
#pragma unroll
        for (int wv = 0; wv < 4; ++wv) {
            float f = __expf(Stats[wv][row][0] - M);
            L   += Stats[wv][row][1] * f;
            val += Ocomb[wv][row * 64 + col] * f;
        }
        int i = i0 + row;
        AV[((size_t)i * BSZ + b) * (NH * DH) + n * DH + col] = (__bf16)(val / L);
    }
}

// ---------------------------------------------------------------------------
// Kernel 3: output GEMM  out[m,m'] = w[m,m'] + sum_h AV[m,h] * o_w[m',h]
// M=4096, N=1024, K=1024; A already bf16.
// ---------------------------------------------------------------------------
__global__ __launch_bounds__(256) void out_gemm_kernel(
    const __bf16* __restrict__ AVm, const float* __restrict__ OW,
    const float* __restrict__ Win, float* __restrict__ Out)
{
    __shared__ __bf16 As[64 * 32];
    __shared__ __bf16 Ws[128 * 32];

    const int tid  = threadIdx.x;
    const int wave = tid >> 5, lane = tid & 31;
    const int hsel = lane >> 4, lrow = lane & 15;
    const int kb = hsel * 8, koff = hsel * 16;
    const int mw = wave & 1, nw = wave >> 1;
    const int m0 = blockIdx.y * 64, n0 = blockIdx.x * 128;

    v8f acc[2][2] = {};

    for (int k0 = 0; k0 < DMODEL; k0 += 32) {
        __syncthreads();
        {   // A tile already bf16: straight 16B copy
            int r = tid >> 2, c = (tid & 3) * 8;
            *(uint4*)(As + r * 32 + c) =
                *(const uint4*)(AVm + (size_t)(m0 + r) * DMODEL + k0 + c);
        }
#pragma unroll
        for (int p = 0; p < 2; ++p) {
            int r = p * 64 + (tid >> 2), c = (tid & 3) * 8;
            cvt8(OW + (size_t)(n0 + r) * DMODEL + k0 + c, Ws + r * 32 + c);
        }
        __syncthreads();

        Frag16 af[2], bf[2];
#pragma unroll
        for (int mi = 0; mi < 2; ++mi) {
            const __bf16* p = As + (mw * 32 + mi * 16 + lrow) * 32;
            af[mi].q[0] = *(const uint4*)(p + kb);
            af[mi].q[1] = *(const uint4*)(p + kb + 16);
        }
#pragma unroll
        for (int ni = 0; ni < 2; ++ni) {
            const __bf16* p = Ws + (nw * 32 + ni * 16 + lrow) * 32;
            bf[ni].q[0] = *(const uint4*)(p + kb);
            bf[ni].q[1] = *(const uint4*)(p + kb + 16);
        }
#pragma unroll
        for (int mi = 0; mi < 2; ++mi)
#pragma unroll
            for (int ni = 0; ni < 2; ++ni)
                acc[mi][ni] = wmma_bf16(af[mi], bf[ni], acc[mi][ni]);
    }

#pragma unroll
    for (int mi = 0; mi < 2; ++mi)
#pragma unroll
    for (int ni = 0; ni < 2; ++ni)
#pragma unroll
    for (int e = 0; e < 8; ++e) {
        int gm = m0 + mw * 32 + mi * 16 + e + 8 * hsel;
        int gn = n0 + nw * 32 + ni * 16 + lrow;
        size_t idx = (size_t)gm * DMODEL + gn;
        Out[idx] = Win[idx] + acc[mi][ni][e];
    }
}

// ---------------------------------------------------------------------------
// Launch
// ---------------------------------------------------------------------------
extern "C" void kernel_launch(void* const* d_in, const int* in_sizes, int n_in,
                              void* d_out, int out_size, void* d_ws, size_t ws_size,
                              hipStream_t stream) {
    const float* w    = (const float*)d_in[0];   // [1024,4,1024]
    const float* r    = (const float*)d_in[1];   // [1024,1024]
    const float* qkvw = (const float*)d_in[2];   // [3072,1024]
    const float* rw   = (const float*)d_in[3];   // [1024,1024]
    const float* ow   = (const float*)d_in[4];   // [1024,1024]
    const float* rrb  = (const float*)d_in[5];   // [16,64]
    const float* rwb  = (const float*)d_in[6];   // [16,64]

    char* ws = (char*)d_ws;
    const size_t MB = 1024u * 1024u;
    __bf16* Qw = (__bf16*)(ws + 0 * MB);    // [b,n,i,d]  8 MB
    __bf16* Qr = (__bf16*)(ws + 8 * MB);    // [b,n,i,d]  8 MB
    __bf16* Kt = (__bf16*)(ws + 16 * MB);   // [b,n,j,d]  8 MB
    __bf16* Vt = (__bf16*)(ws + 24 * MB);   // [b,n,d,j]  8 MB
    __bf16* RK = (__bf16*)(ws + 32 * MB);   // [n,j,d]    2 MB
    __bf16* AV = (__bf16*)(ws + 34 * MB);   // [(i,b),h]  8 MB

    // 1) QKV projection: M=4096, N=3072
    proj_gemm_kernel<<<dim3(24, 64), 256, 0, stream>>>(
        w, qkvw, DMODEL, 0, rwb, rrb, Qw, Qr, Kt, Vt, RK);
    // 2) positional projection: M=1024, N=1024
    proj_gemm_kernel<<<dim3(8, 16), 256, 0, stream>>>(
        r, rw, DMODEL, 1, rwb, rrb, Qw, Qr, Kt, Vt, RK);
    // 3) fused relative attention
    attn_kernel<<<dim3(64, 4, 16), 128, 0, stream>>>(Qw, Qr, Kt, Vt, RK, AV);
    // 4) output projection + residual
    out_gemm_kernel<<<dim3(8, 64), 256, 0, stream>>>(AV, ow, w, (float*)d_out);
}
